// NcmClassifier_83270825935352
// MI455X (gfx1250) — matrix operand order repaired and checked by reading, not compiled, not run on hardware
//
#include <hip/hip_runtime.h>
#include <math.h>

typedef __attribute__((ext_vector_type(4)))  unsigned int v4u;
typedef __attribute__((ext_vector_type(8)))  float        v8f;
typedef __attribute__((ext_vector_type(16))) __bf16       v16bf;

#define BM 128
#define BN 128
#define BK 32
#define LDK 40   // padded LDS row stride in bf16 elements (80 bytes, keeps 16B alignment)

union FragU { v4u u[2]; v16bf b; };

// A-operand fragment (16x32 bf16): lanes 0-15 get K 0-7 (v0-3) + K 16-23 (v4-7),
// lanes 16-31 get K 8-15 + K 24-31.  p already includes row*LDK + (lane>>4)*8.
static __device__ __forceinline__ v16bf ld_fragA(const unsigned short* p) {
  FragU f;
  f.u[0] = *(const v4u*)(p);        // 16 bytes: K chunk 1
  f.u[1] = *(const v4u*)(p + 16);   // +32 bytes: K chunk 2
  return f.b;
}

// B-operand fragment (32x16 bf16, K contiguous per column): lanes 0-15 get K 0-7 + K 16-23,
// lanes 16-31 get K 8-15 + K 24-31.  p already includes row*LDK + (lane>>4)*8.
static __device__ __forceinline__ v16bf ld_fragB(const unsigned short* p) {
  FragU f;
  f.u[0] = *(const v4u*)(p);        // 16 bytes: K chunk 1
  f.u[1] = *(const v4u*)(p + 16);   // +32 bytes: K chunk 2
  return f.b;
}

static __device__ __forceinline__ v8f wmma_bf16(v16bf a, v16bf b, v8f c) {
  // (neg_a, A, neg_b, B, c_mod, C, reuse_a, reuse_b)
  return __builtin_amdgcn_wmma_f32_16x16x32_bf16(false, a, false, b, (short)0, c,
                                                 false, false);
}

// fp32 -> bf16 hi + bf16 residual (bf16x3 split-precision trick)
static __device__ __forceinline__ void split_bf16(float f, unsigned short& h,
                                                  unsigned short& l) {
  __bf16 hb = (__bf16)f;
  float  hf = (float)hb;
  __bf16 lb = (__bf16)(f - hf);
  h = __builtin_bit_cast(unsigned short, hb);
  l = __builtin_bit_cast(unsigned short, lb);
}

__global__ void recip_kernel(const float* __restrict__ T, float* __restrict__ rT, int D) {
  int i = blockIdx.x * blockDim.x + threadIdx.x;
  if (i < D) rT[i] = 1.0f / T[i];
}

// One-time split of a row-major [nrows, D] fp32 matrix (scaled by rT) into bf16 hi/lo.
// 4 elements per thread, bandwidth bound.
__global__ void split_kernel(const float* __restrict__ M, const float* __restrict__ rT,
                             unsigned short* __restrict__ Hi,
                             unsigned short* __restrict__ Lo,
                             long long total, int D) {
  long long i = ((long long)blockIdx.x * blockDim.x + threadIdx.x) * 4;
  if (i >= total) return;
  float4 mv = *(const float4*)(M + i);
  int d = (int)(i % D);                 // D multiple of 4, so 4 elems share a row
  float4 tv = *(const float4*)(rT + d);
  float f0 = mv.x * tv.x, f1 = mv.y * tv.y, f2 = mv.z * tv.z, f3 = mv.w * tv.w;
  unsigned short h0, h1, h2, h3, l0, l1, l2, l3;
  split_bf16(f0, h0, l0); split_bf16(f1, h1, l1);
  split_bf16(f2, h2, l2); split_bf16(f3, h3, l3);
  uint2 hp, lp;
  hp.x = (unsigned)h0 | ((unsigned)h1 << 16);
  hp.y = (unsigned)h2 | ((unsigned)h3 << 16);
  lp.x = (unsigned)l0 | ((unsigned)l1 << 16);
  lp.y = (unsigned)l2 | ((unsigned)l3 << 16);
  *(uint2*)(Hi + i) = hp;
  *(uint2*)(Lo + i) = lp;
}

// one wave (32 lanes) per row; coalesced; wave32 xor-shuffle reduction (fp32 accurate)
__global__ void row_norm_kernel(const float* __restrict__ M, const float* __restrict__ rT,
                                float* __restrict__ out, int nrows, int D) {
  int lane = threadIdx.x & 31;
  int wave = threadIdx.x >> 5;
  int row  = blockIdx.x * (blockDim.x >> 5) + wave;
  if (row >= nrows) return;
  float s = 0.0f;
  for (int d = lane; d < D; d += 32) {
    float v = M[(size_t)row * D + d] * rT[d];
    s += v * v;
  }
#pragma unroll
  for (int off = 16; off > 0; off >>= 1) s += __shfl_xor(s, off, 32);
  if (lane == 0) out[row] = s;
}

__global__ __launch_bounds__(256) void ncm_wmma_kernel(
    const unsigned short* __restrict__ Xhi, const unsigned short* __restrict__ Xlo,
    const unsigned short* __restrict__ Chi, const unsigned short* __restrict__ Clo,
    const float* __restrict__ xx, const float* __restrict__ cc,
    float* __restrict__ Out, int B, int C, int D) {
  __shared__ unsigned short sAhi[BM * LDK];
  __shared__ unsigned short sAlo[BM * LDK];
  __shared__ unsigned short sBhi[BN * LDK];
  __shared__ unsigned short sBlo[BN * LDK];

  const int tid   = threadIdx.x;
  const int lane  = tid & 31;
  const int wave  = tid >> 5;
  const int waveM = wave & 3;   // 4 strips of 32 rows
  const int waveN = wave >> 2;  // 2 strips of 64 cols
  const int m0    = blockIdx.y * BM;
  const int n0    = blockIdx.x * BN;
  const int laneR = lane & 15;
  const int laneH = lane >> 4;

  v8f acc[2][4] = {};

  // register double-buffer for the staged tile (2 slots of 16B per array per thread)
  v4u rAhi[2], rAlo[2], rBhi[2], rBlo[2];
  const v4u vzero = {0u, 0u, 0u, 0u};

  auto stage = [&](int ks) {   // global -> registers (issued early, hidden by WMMA)
    int k0 = ks * BK;
#pragma unroll
    for (int s = 0; s < 2; ++s) {
      int idx = tid + s * 256;        // 0..511 slots, 4 k-chunks per row
      int row = idx >> 2;             // 0..127
      int kc  = (idx & 3) << 3;       // 0,8,16,24
      size_t aoff = (size_t)(m0 + row) * D + k0 + kc;
      rAhi[s] = *(const v4u*)(Xhi + aoff);
      rAlo[s] = *(const v4u*)(Xlo + aoff);
      int grow = n0 + row;            // centroid index (WMMA-B column)
      if (grow < C) {
        size_t boff = (size_t)grow * D + k0 + kc;
        rBhi[s] = *(const v4u*)(Chi + boff);
        rBlo[s] = *(const v4u*)(Clo + boff);
      } else {
        rBhi[s] = vzero;
        rBlo[s] = vzero;
      }
    }
  };
  auto commit = [&]() {        // registers -> LDS
#pragma unroll
    for (int s = 0; s < 2; ++s) {
      int idx = tid + s * 256;
      int row = idx >> 2;
      int kc  = (idx & 3) << 3;
      int la  = row * LDK + kc;
      *(v4u*)(&sAhi[la]) = rAhi[s];
      *(v4u*)(&sAlo[la]) = rAlo[s];
      *(v4u*)(&sBhi[la]) = rBhi[s];
      *(v4u*)(&sBlo[la]) = rBlo[s];
    }
  };

  stage(0);
  commit();
  __syncthreads();

  const int nk = D / BK;
  for (int ks = 0; ks < nk; ++ks) {
    const bool more = (ks + 1) < nk;
    if (more) stage(ks + 1);   // global loads in flight during the WMMA block

    // ---- WMMA: 2(M) x 4(N) tiles, 3 products each (bf16x3) ----
    v16bf aHi[2], aLo[2];
#pragma unroll
    for (int mi = 0; mi < 2; ++mi) {
      int r = waveM * 32 + mi * 16 + laneR;
      aHi[mi] = ld_fragA(&sAhi[r * LDK + laneH * 8]);
      aLo[mi] = ld_fragA(&sAlo[r * LDK + laneH * 8]);
    }
#pragma unroll
    for (int ni = 0; ni < 4; ++ni) {
      int r = waveN * 64 + ni * 16 + laneR;
      v16bf bHi = ld_fragB(&sBhi[r * LDK + laneH * 8]);
      v16bf bLo = ld_fragB(&sBlo[r * LDK + laneH * 8]);
#pragma unroll
      for (int mi = 0; mi < 2; ++mi) {
        acc[mi][ni] = wmma_bf16(aHi[mi], bHi, acc[mi][ni]);
        acc[mi][ni] = wmma_bf16(aLo[mi], bHi, acc[mi][ni]);
        acc[mi][ni] = wmma_bf16(aHi[mi], bLo, acc[mi][ni]);
      }
    }

    __syncthreads();           // all lanes done reading this LDS tile
    if (more) {
      commit();                // overwrite LDS with prefetched tile
      __syncthreads();         // visible before next compute
    }
  }

  // ---- fused epilogue: out = -0.5*sqrt(max(xx - 2*cross + cc, 0)) ----
  // C/D layout: lane%16 = N, VGPR r -> M = r + 8*(lane>=16)
#pragma unroll
  for (int mi = 0; mi < 2; ++mi) {
    int rowBase = m0 + waveM * 32 + mi * 16 + laneH * 8;
    float xxv[8];
#pragma unroll
    for (int r = 0; r < 8; ++r) xxv[r] = xx[rowBase + r];
#pragma unroll
    for (int ni = 0; ni < 4; ++ni) {
      int col = n0 + waveN * 64 + ni * 16 + laneR;
      if (col < C) {
        float ccv = cc[col];
#pragma unroll
        for (int r = 0; r < 8; ++r) {
          float d2 = xxv[r] - 2.0f * acc[mi][ni][r] + ccv;
          d2 = fmaxf(d2, 0.0f);
          Out[(size_t)(rowBase + r) * C + col] = -0.5f * sqrtf(d2);
        }
      }
    }
  }
}

extern "C" void kernel_launch(void* const* d_in, const int* in_sizes, int n_in,
                              void* d_out, int out_size, void* d_ws, size_t ws_size,
                              hipStream_t stream) {
  const float* X    = (const float*)d_in[0];  // [B, D]
  const float* Cent = (const float*)d_in[1];  // [C, D]
  // d_in[2] = stddevs: unused by the reference computation
  const float* T    = (const float*)d_in[3];  // [D]
  float* Out = (float*)d_out;                 // [B, C]

  const int D = in_sizes[3];
  const int B = in_sizes[0] / D;
  const int C = in_sizes[1] / D;

  // workspace layout: xx[B] | cc[C] | rT[D] | Xhi | Xlo | Chi | Clo (bf16 arrays)
  // needs ws_size >= (B+C+D)*4 + (2*B*D + 2*C*D)*2 bytes  (~35.8 MB here)
  float* xx = (float*)d_ws;
  float* cc = xx + B;
  float* rT = cc + C;
  size_t off = ((size_t)(B + C + D) * sizeof(float) + 15) & ~(size_t)15;
  unsigned short* Xhi = (unsigned short*)((char*)d_ws + off);
  unsigned short* Xlo = Xhi + (size_t)B * D;
  unsigned short* Chi = Xlo + (size_t)B * D;
  unsigned short* Clo = Chi + (size_t)C * D;

  recip_kernel<<<(D + 255) / 256, 256, 0, stream>>>(T, rT, D);

  long long totX = (long long)B * D, totC = (long long)C * D;
  split_kernel<<<(unsigned)((totX / 4 + 255) / 256), 256, 0, stream>>>(X, rT, Xhi, Xlo, totX, D);
  split_kernel<<<(unsigned)((totC / 4 + 255) / 256), 256, 0, stream>>>(Cent, rT, Chi, Clo, totC, D);

  row_norm_kernel<<<(B + 7) / 8, 256, 0, stream>>>(X, rT, xx, B, D);
  row_norm_kernel<<<(C + 7) / 8, 256, 0, stream>>>(Cent, rT, cc, C, D);

  dim3 grid((C + BN - 1) / BN, (B + BM - 1) / BM);
  ncm_wmma_kernel<<<grid, 256, 0, stream>>>(Xhi, Xlo, Chi, Clo, xx, cc, Out, B, C, D);
}
